// BlocDiagLinear_67680094651005
// MI455X (gfx1250) — matrix operand order repaired and checked by reading, not compiled, not run on hardware
//
#include <hip/hip_runtime.h>

// ---------------------------------------------------------------------------
// Block-diagonal linear:  out[b, n*128 + r] = sum_c x[b, n*128 + c] * W[n, r, c]
// fp32 in/out, computed with V_WMMA_F32_16X16X32_BF16 using a hi/lo bf16 split
// (3 products: hi*hi + lo*hi + hi*lo  -> ~fp32 accuracy, bandwidth-bound).
// ---------------------------------------------------------------------------

typedef __attribute__((ext_vector_type(16))) __bf16 v16bf;
typedef __attribute__((ext_vector_type(8)))  __bf16 v8bf;   // 16B, matches ds_load_b128
typedef __attribute__((ext_vector_type(2)))  __bf16 v2bf;   // 4B packed store
typedef __attribute__((ext_vector_type(8)))  float  v8f;

#define NBLK        64
#define BR          128
#define BC          128
#define BATCH       4096
#define ROWS_PER_WG 128
#define LDS_STRIDE  136   // 128 + 8 bf16 pad -> 272B row stride, conflict-free b128 reads

__global__ __launch_bounds__(256)
void bdl_wmma_kernel(const float* __restrict__ x,
                     const float* __restrict__ blocks,
                     float* __restrict__ out)
{
    // Pre-converted weights for this block, bf16 hi + residual lo (68 KB total).
    __shared__ __bf16 lds_hi[BR * LDS_STRIDE];
    __shared__ __bf16 lds_lo[BR * LDS_STRIDE];

    const int n   = blockIdx.x;   // block id 0..63
    const int tm  = blockIdx.y;   // batch tile 0..31
    const int tid = threadIdx.x;

    // ---------------- Phase 1: stage W[n] into LDS as bf16 hi/lo ------------
    // Coalesced float2 loads; packed 4B bf16-pair stores.
    const float* wsrc = blocks + (size_t)n * (BR * BC);
    for (int base = tid * 2; base < BR * BC; base += 256 * 2) {
        const int r = base >> 7;       // /128
        const int c = base & 127;
        float2 f = *(const float2*)(wsrc + base);
        __bf16 h0 = (__bf16)f.x;
        __bf16 h1 = (__bf16)f.y;
        __bf16 l0 = (__bf16)(f.x - (float)h0);
        __bf16 l1 = (__bf16)(f.y - (float)h1);
        v2bf hp; hp[0] = h0; hp[1] = h1;
        v2bf lp; lp[0] = l0; lp[1] = l1;
        *(v2bf*)&lds_hi[r * LDS_STRIDE + c] = hp;   // c is even -> 4B aligned
        *(v2bf*)&lds_lo[r * LDS_STRIDE + c] = lp;
    }
    __syncthreads();

    // ---------------- Phase 2: WMMA main loop -------------------------------
    const int wave   = tid >> 5;        // 0..7 (wave32)
    const int lane   = tid & 31;
    const int lane16 = lane & 15;
    const int laneHi = lane >> 4;       // 0 or 1

    const int m0 = tm * ROWS_PER_WG + wave * 16;   // this wave's 16 batch rows
    // A-fragment base: row m0+lane16, column window starts at laneHi*8
    // (16-bit A 16x32 layout: lanes 0-15 hold K {0..7,16..23}, lanes 16-31 {8..15,24..31})
    const float* xrow = x + (size_t)(m0 + lane16) * (NBLK * BC)
                          + (size_t)n * BC + laneHi * 8;

    v8f acc[8] = {};   // 8 N-tiles of 16x16 f32 accumulators

    #pragma unroll
    for (int kk = 0; kk < 4; ++kk) {
        const int k0 = kk * 32;

        // ---- load + split A fragment (x rows) ----
        const float4* ap = (const float4*)(xrow + k0);
        float4 f0 = ap[0];   // K = k0 + base + 0..3
        float4 f1 = ap[1];   // K = k0 + base + 4..7
        float4 f2 = ap[4];   // K = k0 + base + 16..19
        float4 f3 = ap[5];   // K = k0 + base + 20..23
        float af[16] = { f0.x, f0.y, f0.z, f0.w,  f1.x, f1.y, f1.z, f1.w,
                         f2.x, f2.y, f2.z, f2.w,  f3.x, f3.y, f3.z, f3.w };
        v16bf a_hi, a_lo;
        #pragma unroll
        for (int i = 0; i < 16; ++i) {
            __bf16 h = (__bf16)af[i];
            a_hi[i] = h;
            a_lo[i] = (__bf16)(af[i] - (float)h);
        }

        // ---- sweep the 8 N-tiles, B fragments from LDS ----
        #pragma unroll
        for (int nt = 0; nt < 8; ++nt) {
            // B 32x16 layout: N = lane16, lanes 0-15 hold K {0..7,16..23}, lanes 16-31 {8..15,24..31}
            const int r    = nt * 16 + lane16;               // output column
            const int coff = k0 + laneHi * 8;                // K base for this lane half
            const __bf16* bh = &lds_hi[r * LDS_STRIDE + coff];
            const __bf16* bl = &lds_lo[r * LDS_STRIDE + coff];
            v8bf bh0 = *(const v8bf*)(bh);       // 16B-aligned ds_load_b128
            v8bf bh1 = *(const v8bf*)(bh + 16);
            v8bf bl0 = *(const v8bf*)(bl);
            v8bf bl1 = *(const v8bf*)(bl + 16);
            v16bf b_hi, b_lo;
            #pragma unroll
            for (int i = 0; i < 8; ++i) {
                b_hi[i] = bh0[i]; b_hi[8 + i] = bh1[i];
                b_lo[i] = bl0[i]; b_lo[8 + i] = bl1[i];
            }

            // D = A*B + C : hi*hi + lo*hi + hi*lo  (drop lo*lo, ~2^-16 residual)
            acc[nt] = __builtin_amdgcn_wmma_f32_16x16x32_bf16(
                          false, a_hi, false, b_hi, (short)0, acc[nt], false, false);
            acc[nt] = __builtin_amdgcn_wmma_f32_16x16x32_bf16(
                          false, a_lo, false, b_hi, (short)0, acc[nt], false, false);
            acc[nt] = __builtin_amdgcn_wmma_f32_16x16x32_bf16(
                          false, a_hi, false, b_lo, (short)0, acc[nt], false, false);
        }
    }

    // ---------------- store: C/D layout M = laneHi*8 + vgpr, N = lane16 -----
    float* obase = out + (size_t)m0 * (NBLK * BR) + (size_t)n * BR;
    #pragma unroll
    for (int nt = 0; nt < 8; ++nt) {
        #pragma unroll
        for (int j = 0; j < 8; ++j) {
            const int row = laneHi * 8 + j;
            obase[(size_t)row * (NBLK * BR) + nt * 16 + lane16] = acc[nt][j];
        }
    }
}

extern "C" void kernel_launch(void* const* d_in, const int* in_sizes, int n_in,
                              void* d_out, int out_size, void* d_ws, size_t ws_size,
                              hipStream_t stream) {
    const float* x      = (const float*)d_in[0];   // (4096, 64*128) f32
    const float* blocks = (const float*)d_in[1];   // (64, 128, 128) f32
    float* out          = (float*)d_out;           // (4096, 64*128) f32

    dim3 grid(NBLK, BATCH / ROWS_PER_WG);          // (64, 32)
    bdl_wmma_kernel<<<grid, 256, 0, stream>>>(x, blocks, out);
}
